// EncoderLayer_34978213658634
// MI455X (gfx1250) — matrix operand order repaired and checked by reading, not compiled, not run on hardware
//
#include <hip/hip_runtime.h>
#include <cstdint>
#include <cstddef>

#define D_MODEL 1024
#define N_HEAD  16
#define D_HEAD  64
#define D_FF    4096
#define BATCH   2
#define SEQ     2048
#define TOKENS  (BATCH*SEQ)

typedef __attribute__((ext_vector_type(16))) __bf16 v16bf;
typedef __attribute__((ext_vector_type(8)))  float  v8f;

union AF { v16bf v; unsigned u[8]; };

__device__ __forceinline__ unsigned short f2bf(float f) {
  union { float f; unsigned u; } c; c.f = f;
  unsigned r = c.u + 0x7fffu + ((c.u >> 16) & 1u);   // round-to-nearest-even
  return (unsigned short)(r >> 16);
}

// CDNA5 async copy: global -> LDS, 16B per lane, tracked by ASYNCcnt.
// LDS operand = low 32 bits of the generic pointer (flat LDS aperture keeps the
// byte offset in addr[31:0], per CDNA5 ISA aperture rules).
__device__ __forceinline__ void async_copy_b128(const unsigned short* g, unsigned short* l) {
  unsigned lds = (unsigned)(size_t)l;
  unsigned long long ga = (unsigned long long)(size_t)g;
  asm volatile("global_load_async_to_lds_b128 %0, %1, off" :: "v"(lds), "v"(ga) : "memory");
}
__device__ __forceinline__ void wait_async_le1() {
  asm volatile("s_wait_asynccnt 0x1" ::: "memory");
}
__device__ __forceinline__ void wait_async_0() {
  asm volatile("s_wait_asynccnt 0x0" ::: "memory");
}

// ---------------------------------------------------------------- conversions
__global__ __launch_bounds__(256) void cvt_bf16(const float* __restrict__ X,
                                                unsigned short* __restrict__ Y, int n) {
  int i = blockIdx.x * 256 + threadIdx.x;
  if (i < n) Y[i] = f2bf(X[i]);
}

// W[K][N] fp32  ->  WT[N][K] bf16  (so GEMM B-fragments load contiguous K pairs)
__global__ __launch_bounds__(256) void transpose_cvt(const float* __restrict__ W,
                                                     unsigned short* __restrict__ WT,
                                                     int K, int N) {
  int idx = blockIdx.x * 256 + threadIdx.x;
  if (idx >= K * N) return;
  int k = idx / N, n = idx % N;
  WT[(size_t)n * K + k] = f2bf(W[idx]);
}

// ---------------------------------------------------------------- WMMA GEMM
// A[M][K] bf16 row-major, BT[N][K] bf16 (transposed weights), bias[N] f32.
// Block tile: 64 rows x 256 cols (8 waves). Wave tile: 32x64 -> acc[2][4];
// each B fragment feeds two WMMAs. A tile (64x32) is staged to LDS with
// global_load_async_to_lds_b128, double-buffered on ASYNCcnt.
// mode: 0 = bf16 [M][N] (optional ReLU)   1 = bf16 per-head Q/K  [B,H,S,dh]
//       2 = bf16 transposed V [B,H,dh,S]  3 = f32  [M][N]
#define AROW 40   // LDS row stride (elements): 80B rows -> 16B aligned, conflict-free
__global__ __launch_bounds__(256) void gemm_wmma(
    const unsigned short* __restrict__ A, const unsigned short* __restrict__ BT,
    const float* __restrict__ bias, void* __restrict__ outp,
    int M, int K, int N, int mode, int relu) {
  __shared__ unsigned short abuf[2][64 * AROW];
  const int tid  = threadIdx.x;
  const int lane = tid & 31;
  const int wave = tid >> 5;
  const int l15  = lane & 15;
  const int g    = lane >> 4;
  const int mBlk = blockIdx.x * 64;
  const int n0   = blockIdx.y * 256 + ((wave & 3) << 6);
  const int mw   = 32 * (wave >> 2);          // wave's row offset inside block tile

  int akoff[8];
#pragma unroll
  for (int v = 0; v < 8; ++v)
    akoff[v] = 16 * g + 2 * v;

  // async A-tile staging: one 16B chunk per thread covers 64 rows x 32 k
  const int arow = tid >> 2;                  // 0..63
  const int asub = (tid & 3) * 8;             // element offset within k-tile
  const unsigned short* aSrc = A + (size_t)(mBlk + arow) * K + asub;
  unsigned short* aDst0 = &abuf[0][arow * AROW + asub];
  unsigned short* aDst1 = &abuf[1][arow * AROW + asub];

  const unsigned short* bRow[4];
#pragma unroll
  for (int t = 0; t < 4; ++t)
    bRow[t] = BT + (size_t)(n0 + 16 * t + l15) * K + 16 * g;

  v8f acc[2][4];
#pragma unroll
  for (int h = 0; h < 2; ++h)
#pragma unroll
    for (int t = 0; t < 4; ++t) acc[h][t] = (v8f){0.f,0.f,0.f,0.f,0.f,0.f,0.f,0.f};

  async_copy_b128(aSrc, aDst0);               // prologue: fill buffer 0

  for (int k0 = 0; k0 < K; k0 += 32) {
    const int cur = (k0 >> 5) & 1;
    if (k0 + 32 < K) {
      async_copy_b128(aSrc + k0 + 32, cur ? aDst0 : aDst1);
      wait_async_le1();                       // current buffer landed (in-order)
    } else {
      wait_async_0();
    }
    __syncthreads();                          // tile visible to all 8 waves

    AF a0, a1;
    const unsigned short* ab = abuf[cur];
#pragma unroll
    for (int v = 0; v < 8; ++v) {
      a0.u[v] = *(const unsigned*)(ab + (mw + l15) * AROW + akoff[v]);
      a1.u[v] = *(const unsigned*)(ab + (mw + 16 + l15) * AROW + akoff[v]);
    }
#pragma unroll
    for (int t = 0; t < 4; ++t) {
      if (k0 + 256 < K) __builtin_prefetch(bRow[t] + k0 + 256, 0, 1);
      AF b;
#pragma unroll
      for (int v = 0; v < 8; ++v)
        b.u[v] = *(const unsigned*)(bRow[t] + k0 + 2 * v);
      acc[0][t] = __builtin_amdgcn_wmma_f32_16x16x32_bf16(
          false, a0.v, false, b.v, (short)0, acc[0][t], false, false);
      acc[1][t] = __builtin_amdgcn_wmma_f32_16x16x32_bf16(
          false, a1.v, false, b.v, (short)0, acc[1][t], false, false);
    }
    __syncthreads();                          // reads done before next async writes
  }

#pragma unroll
  for (int h = 0; h < 2; ++h) {
    const int mbase = mBlk + mw + 16 * h;
#pragma unroll
    for (int t = 0; t < 4; ++t) {
      const int col = n0 + 16 * t + l15;
      const float bv = bias[col];
#pragma unroll
      for (int r = 0; r < 8; ++r) {
        const int row = mbase + r + 8 * g;
        float val = acc[h][t][r] + bv;
        if (relu) val = fmaxf(val, 0.f);
        if (mode == 0) {
          ((unsigned short*)outp)[(size_t)row * N + col] = f2bf(val);
        } else if (mode == 1) {
          int b = row >> 11, s = row & 2047, hh = col >> 6, d = col & 63;
          ((unsigned short*)outp)[(((size_t)(b * N_HEAD + hh) * SEQ) + s) * D_HEAD + d] = f2bf(val);
        } else if (mode == 2) {
          int b = row >> 11, s = row & 2047, hh = col >> 6, d = col & 63;
          ((unsigned short*)outp)[((size_t)(b * N_HEAD + hh) * D_HEAD + d) * SEQ + s] = f2bf(val);
        } else {
          ((float*)outp)[(size_t)row * N + col] = val;
        }
      }
    }
  }
}

// ---------------------------------------------------------------- attention
// One wave per 16-query tile; flash-style online softmax over K in chunks of 32.
// Q,K: [B,H,S,64] bf16.  Vt: [B,H,64,S] bf16.  out: [B*S, D_MODEL] bf16 (heads concat).
__global__ __launch_bounds__(128) void attention_wmma(
    const unsigned short* __restrict__ Q, const unsigned short* __restrict__ Km,
    const unsigned short* __restrict__ Vt, const float* __restrict__ mask,
    unsigned short* __restrict__ out) {
  __shared__ unsigned short pb[4][16 * 32];   // per-wave P-tile staging (C->A layout)
  const int lane = threadIdx.x & 31;
  const int wave = threadIdx.x >> 5;
  const int l15  = lane & 15;
  const int g    = lane >> 4;
  const int bh   = blockIdx.y;
  const int b    = bh >> 4, h = bh & 15;
  const int s0   = blockIdx.x * 64 + wave * 16;

  int akoff[8];
#pragma unroll
  for (int v = 0; v < 8; ++v)
    akoff[v] = 16 * g + 2 * v;

  const unsigned short* qRow = Q + ((size_t)bh * SEQ + s0 + l15) * D_HEAD;
  AF aq0, aq1;
#pragma unroll
  for (int v = 0; v < 8; ++v) {
    aq0.u[v] = *(const unsigned*)(qRow + akoff[v]);
    aq1.u[v] = *(const unsigned*)(qRow + 32 + akoff[v]);
  }

  v8f oa[4];
#pragma unroll
  for (int t = 0; t < 4; ++t) oa[t] = (v8f){0.f,0.f,0.f,0.f,0.f,0.f,0.f,0.f};
  float mr[8], lr[8];
#pragma unroll
  for (int r = 0; r < 8; ++r) { mr[r] = -3.0e38f; lr[r] = 0.f; }

  const float* maskb = mask + (size_t)b * SEQ;
  unsigned short* pw = pb[wave];

  for (int kc = 0; kc < SEQ; kc += 32) {
    const unsigned short* k0p = Km + ((size_t)bh * SEQ + kc + l15) * D_HEAD + 16 * g;
    const unsigned short* k1p = k0p + 16 * D_HEAD;

    v8f c0 = (v8f){0.f,0.f,0.f,0.f,0.f,0.f,0.f,0.f};
    v8f c1 = (v8f){0.f,0.f,0.f,0.f,0.f,0.f,0.f,0.f};
    AF bk;
#pragma unroll
    for (int v = 0; v < 8; ++v) bk.u[v] = *(const unsigned*)(k0p + 2 * v);
    c0 = __builtin_amdgcn_wmma_f32_16x16x32_bf16(false, aq0.v, false, bk.v, (short)0, c0, false, false);
#pragma unroll
    for (int v = 0; v < 8; ++v) bk.u[v] = *(const unsigned*)(k0p + 32 + 2 * v);
    c0 = __builtin_amdgcn_wmma_f32_16x16x32_bf16(false, aq1.v, false, bk.v, (short)0, c0, false, false);
#pragma unroll
    for (int v = 0; v < 8; ++v) bk.u[v] = *(const unsigned*)(k1p + 2 * v);
    c1 = __builtin_amdgcn_wmma_f32_16x16x32_bf16(false, aq0.v, false, bk.v, (short)0, c1, false, false);
#pragma unroll
    for (int v = 0; v < 8; ++v) bk.u[v] = *(const unsigned*)(k1p + 32 + 2 * v);
    c1 = __builtin_amdgcn_wmma_f32_16x16x32_bf16(false, aq1.v, false, bk.v, (short)0, c1, false, false);

    const float msk0 = maskb[kc + l15];
    const float msk1 = maskb[kc + 16 + l15];

#pragma unroll
    for (int r = 0; r < 8; ++r) {
      float sA = c0[r] * 0.125f + msk0;   // 1/sqrt(64)
      float sB = c1[r] * 0.125f + msk1;
      float mx = fmaxf(sA, sB);
#pragma unroll
      for (int d = 1; d < 16; d <<= 1) mx = fmaxf(mx, __shfl_xor(mx, d, 32));
      float mnew  = fmaxf(mr[r], mx);
      float alpha = __expf(mr[r] - mnew);
      float pA = __expf(sA - mnew);
      float pB = __expf(sB - mnew);
      float sum = pA + pB;
#pragma unroll
      for (int d = 1; d < 16; d <<= 1) sum += __shfl_xor(sum, d, 32);
      lr[r] = lr[r] * alpha + sum;
      mr[r] = mnew;
      oa[0][r] *= alpha; oa[1][r] *= alpha; oa[2][r] *= alpha; oa[3][r] *= alpha;
      const int row = r + 8 * g;            // C-layout row for this lane half
      pw[row * 32 + l15]      = f2bf(pA);
      pw[row * 32 + 16 + l15] = f2bf(pB);
    }

    AF ap;                                  // P tile re-read in A-fragment layout
#pragma unroll
    for (int v = 0; v < 8; ++v)
      ap.u[v] = *(const unsigned*)(pw + l15 * 32 + akoff[v]);

#pragma unroll
    for (int t = 0; t < 4; ++t) {
      const unsigned short* vp = Vt + ((size_t)bh * D_HEAD + 16 * t + l15) * SEQ + kc + 16 * g;
      AF bv;
#pragma unroll
      for (int v = 0; v < 8; ++v) bv.u[v] = *(const unsigned*)(vp + 2 * v);
      oa[t] = __builtin_amdgcn_wmma_f32_16x16x32_bf16(
          false, ap.v, false, bv.v, (short)0, oa[t], false, false);
    }
  }

#pragma unroll
  for (int r = 0; r < 8; ++r) {
    const float inv  = 1.0f / lr[r];
    const int   srow = s0 + r + 8 * g;
    const size_t base = ((size_t)(b * SEQ) + srow) * D_MODEL + h * D_HEAD;
#pragma unroll
    for (int t = 0; t < 4; ++t)
      out[base + 16 * t + l15] = f2bf(oa[t][r] * inv);
  }
}

// ---------------------------------------------------------------- residual + LayerNorm
__global__ __launch_bounds__(256) void add_ln(
    const float* __restrict__ resid, const float* __restrict__ y,
    const float* __restrict__ gam, const float* __restrict__ bet,
    float* __restrict__ outf, unsigned short* __restrict__ outb, int writeB16) {
  __shared__ float s1[256], s2[256];
  const int row = blockIdx.x, tid = threadIdx.x;
  const float* rp = resid + (size_t)row * D_MODEL;
  const float* yp = y     + (size_t)row * D_MODEL;
  float v[4]; float sum = 0.f, sq = 0.f;
#pragma unroll
  for (int j = 0; j < 4; ++j) {
    int i = tid + j * 256;
    float t = rp[i] + yp[i];
    v[j] = t; sum += t; sq += t * t;
  }
  s1[tid] = sum; s2[tid] = sq; __syncthreads();
  for (int st = 128; st > 0; st >>= 1) {
    if (tid < st) { s1[tid] += s1[tid + st]; s2[tid] += s2[tid + st]; }
    __syncthreads();
  }
  const float mu   = s1[0] * (1.0f / D_MODEL);
  const float var  = s2[0] * (1.0f / D_MODEL) - mu * mu;
  const float rstd = rsqrtf(var + 1e-5f);
  float* op = outf + (size_t)row * D_MODEL;
#pragma unroll
  for (int j = 0; j < 4; ++j) {
    int i = tid + j * 256;
    float o = (v[j] - mu) * rstd * gam[i] + bet[i];
    op[i] = o;
    if (writeB16) outb[(size_t)row * D_MODEL + i] = f2bf(o);
  }
}

// ---------------------------------------------------------------- launch
extern "C" void kernel_launch(void* const* d_in, const int* in_sizes, int n_in,
                              void* d_out, int out_size, void* d_ws, size_t ws_size,
                              hipStream_t stream) {
  const float* x     = (const float*)d_in[0];
  const float* mask  = (const float*)d_in[1];
  const float* w_q   = (const float*)d_in[2];  const float* b_q   = (const float*)d_in[3];
  const float* w_k   = (const float*)d_in[4];  const float* b_k   = (const float*)d_in[5];
  const float* w_v   = (const float*)d_in[6];  const float* b_v   = (const float*)d_in[7];
  const float* w_o   = (const float*)d_in[8];  const float* b_o   = (const float*)d_in[9];
  const float* w_ff1 = (const float*)d_in[10]; const float* b_ff1 = (const float*)d_in[11];
  const float* w_ff2 = (const float*)d_in[12]; const float* b_ff2 = (const float*)d_in[13];
  const float* ln1_g = (const float*)d_in[14]; const float* ln1_b = (const float*)d_in[15];
  const float* ln2_g = (const float*)d_in[16]; const float* ln2_b = (const float*)d_in[17];

  char* ws = (char*)d_ws;
  size_t off = 0;
  auto alloc = [&](size_t bytes) -> void* {
    void* p = ws + off;
    off += (bytes + 255) & ~((size_t)255);
    return p;
  };

  unsigned short* xb   = (unsigned short*)alloc((size_t)TOKENS * D_MODEL * 2);
  unsigned short* wqT  = (unsigned short*)alloc((size_t)D_MODEL * D_MODEL * 2);
  unsigned short* wkT  = (unsigned short*)alloc((size_t)D_MODEL * D_MODEL * 2);
  unsigned short* wvT  = (unsigned short*)alloc((size_t)D_MODEL * D_MODEL * 2);
  unsigned short* woT  = (unsigned short*)alloc((size_t)D_MODEL * D_MODEL * 2);
  unsigned short* w1T  = (unsigned short*)alloc((size_t)D_FF * D_MODEL * 2);
  unsigned short* w2T  = (unsigned short*)alloc((size_t)D_MODEL * D_FF * 2);
  unsigned short* qb   = (unsigned short*)alloc((size_t)TOKENS * D_MODEL * 2);
  unsigned short* kb   = (unsigned short*)alloc((size_t)TOKENS * D_MODEL * 2);
  unsigned short* vtb  = (unsigned short*)alloc((size_t)TOKENS * D_MODEL * 2);
  unsigned short* attn = (unsigned short*)alloc((size_t)TOKENS * D_MODEL * 2);
  float*          proj = (float*)alloc((size_t)TOKENS * D_MODEL * 4);
  float*          ln1f = (float*)alloc((size_t)TOKENS * D_MODEL * 4);
  unsigned short* ln1h = (unsigned short*)alloc((size_t)TOKENS * D_MODEL * 2);
  // Aliased scratch: FF1 output (32 MB) reuses qb..attn (dead after O-proj);
  // FF2 f32 output reuses proj (dead after first add_ln).
  unsigned short* ff1 = qb;
  float*          ff2 = proj;

  const int NE = TOKENS * D_MODEL;                       // 4M elements
  cvt_bf16<<<NE / 256, 256, 0, stream>>>(x, xb, NE);
  transpose_cvt<<<(D_MODEL * D_MODEL) / 256, 256, 0, stream>>>(w_q,   wqT, D_MODEL, D_MODEL);
  transpose_cvt<<<(D_MODEL * D_MODEL) / 256, 256, 0, stream>>>(w_k,   wkT, D_MODEL, D_MODEL);
  transpose_cvt<<<(D_MODEL * D_MODEL) / 256, 256, 0, stream>>>(w_v,   wvT, D_MODEL, D_MODEL);
  transpose_cvt<<<(D_MODEL * D_MODEL) / 256, 256, 0, stream>>>(w_o,   woT, D_MODEL, D_MODEL);
  transpose_cvt<<<(D_MODEL * D_FF)   / 256, 256, 0, stream>>>(w_ff1, w1T, D_MODEL, D_FF);
  transpose_cvt<<<(D_FF * D_MODEL)   / 256, 256, 0, stream>>>(w_ff2, w2T, D_FF, D_MODEL);

  const dim3 blk(256);
  const dim3 gDM(TOKENS / 64, D_MODEL / 256);
  gemm_wmma<<<gDM, blk, 0, stream>>>(xb, wqT, b_q, qb,  TOKENS, D_MODEL, D_MODEL, 1, 0);
  gemm_wmma<<<gDM, blk, 0, stream>>>(xb, wkT, b_k, kb,  TOKENS, D_MODEL, D_MODEL, 1, 0);
  gemm_wmma<<<gDM, blk, 0, stream>>>(xb, wvT, b_v, vtb, TOKENS, D_MODEL, D_MODEL, 2, 0);

  attention_wmma<<<dim3(SEQ / 64, BATCH * N_HEAD), dim3(128), 0, stream>>>(qb, kb, vtb, mask, attn);

  gemm_wmma<<<gDM, blk, 0, stream>>>(attn, woT, b_o, proj, TOKENS, D_MODEL, D_MODEL, 3, 0);
  add_ln<<<TOKENS, 256, 0, stream>>>(x, proj, ln1_g, ln1_b, ln1f, ln1h, 1);

  gemm_wmma<<<dim3(TOKENS / 64, D_FF / 256), blk, 0, stream>>>(ln1h, w1T, b_ff1, ff1,
                                                               TOKENS, D_MODEL, D_FF, 0, 1);
  gemm_wmma<<<gDM, blk, 0, stream>>>(ff1, w2T, b_ff2, ff2, TOKENS, D_FF, D_MODEL, 3, 0);
  add_ln<<<TOKENS, 256, 0, stream>>>(ln1f, ff2, ln2_g, ln2_b, (float*)d_out, nullptr, 0);
}
